// Sins_75213467288228
// MI455X (gfx1250) — compile-verified, matrix operand order checked by reading, not run on hardware
//
#include <hip/hip_runtime.h>

typedef __attribute__((ext_vector_type(16))) _Float16 v16h;
typedef __attribute__((ext_vector_type(8)))  _Float16 v8h;
typedef __attribute__((ext_vector_type(8)))  float    v8f;

// ---- problem constants ----
constexpr int B_    = 2;
constexpr int NF_   = 1024;
constexpr int BLK   = 512;
constexpr int T_    = NF_ * BLK;        // 524288 = 2^19
constexpr int WIN_  = 2048;
constexpr int BINS_ = WIN_ / 2 + 1;     // 1025
constexpr int NH_   = 128;
constexpr int NMAG_ = 256;
constexpr int NMEL_ = 80;
constexpr int DIN_  = NMEL_ + 2 * BLK;  // 1104
constexpr int DOUT_ = NH_ + BINS_ + 2 * NMAG_; // 1665
constexpr int KP_   = 1120;             // DIN padded to mult of 32
constexpr int NP_   = 1680;             // DOUT padded to mult of 16
constexpr int BT_   = B_ * T_;
constexpr float INV_SR = 1.0f / 44100.0f;
constexpr float TWO_PI = 6.28318530717958647692f;
constexpr float PI_F   = 3.14159265358979323846f;

// ---- workspace layout (float element offsets) ----
constexpr size_t OFF_OFFS = 0;                              // B*NF
constexpr size_t OFF_SINB = 4096;                           // B*T
constexpr size_t OFF_CTRL = OFF_SINB + (size_t)BT_;         // 2048*1665 (stride DOUT_)
constexpr size_t OFF_AMPS = OFF_CTRL + (size_t)2048 * DOUT_;// 2048*128
constexpr size_t OFF_FT   = OFF_AMPS + (size_t)2048 * NH_;  // B*1025*2048
constexpr size_t OFF_YN   = OFF_FT + (size_t)B_ * 1025 * WIN_; // B*1024*1024
constexpr size_t OFF_AH   = OFF_YN + (size_t)B_ * NF_ * 1024;  // halfs: 2048*KP (as float slots /2)
constexpr size_t OFF_WH   = OFF_AH + (size_t)2048 * KP_ / 2;   // halfs: NP*KP

__device__ __forceinline__ float2 cmulf(float2 a, float2 b) {
    return make_float2(a.x * b.x - a.y * b.y, a.x * b.y + a.y * b.x);
}
__device__ __forceinline__ float hannw(int i, float invN) {
    return 0.5f - 0.5f * __cosf(TWO_PI * (float)i * invN);
}

// iterative radix-2 DIT FFT; input must be in bit-reversed positions.
template <int N>
__device__ void fft_lds(float2* buf, int tid, int nthr, float sign) {
    for (int len = 2; len <= N; len <<= 1) {
        int half = len >> 1;
        float ang = sign * TWO_PI / (float)len;
        for (int bf = tid; bf < (N >> 1); bf += nthr) {
            int pos = bf & (half - 1);
            int i0  = ((bf - pos) << 1) + pos;
            int i1  = i0 + half;
            float s, c;
            __sincosf(ang * (float)pos, &s, &c);
            float2 u = buf[i0], v = buf[i1];
            float2 t = make_float2(v.x * c - v.y * s, v.x * s + v.y * c);
            buf[i0] = make_float2(u.x + t.x, u.y + t.y);
            buf[i1] = make_float2(u.x - t.x, u.y - t.y);
        }
        __syncthreads();
    }
}

// ---------- 1: per-frame phase offsets (sequential scan, tiny) ----------
__global__ void k_offsets(const float* __restrict__ f0, float* __restrict__ off) {
    int b = blockIdx.x;
    if (threadIdx.x != 0) return;
    float acc = 0.0f;
    for (int f = 0; f < NF_; ++f) {
        off[b * NF_ + f] = acc;
        float s0 = f0[b * NF_ + f] * INV_SR;
        float s1 = (f < NF_ - 1) ? f0[b * NF_ + f + 1] * INV_SR : s0;
        float ds = s1 - s0;
        float radend = 512.0f * s0 + 255.5f * ds;
        float r2 = radend + 0.5f;
        r2 = r2 - truncf(r2);
        r2 -= 0.5f;
        acc += r2;
        acc = acc - truncf(acc);
    }
}

// ---------- 2: base sinusoid (harmonic #1, GEMM input) ----------
__global__ void k_sinbase(const float* __restrict__ f0, const float* __restrict__ off,
                          float* __restrict__ sinb) {
    int i = blockIdx.x * blockDim.x + threadIdx.x;
    if (i >= BT_) return;
    int b = i >> 19, t = i & (T_ - 1);
    int f = t >> 9, n = t & 511;
    int row = b * NF_ + f;
    float s0 = f0[row] * INV_SR;
    float s1 = (f < NF_ - 1) ? f0[row + 1] * INV_SR : s0;
    float ds = s1 - s0;
    float fn = (float)n;
    float rad = s0 * (fn + 1.0f) + 0.5f * ds * fn * (fn + 1.0f) * (1.0f / 512.0f) + off[row];
    sinb[i] = __sinf(TWO_PI * rad);
}

// ---------- 3/4: pack f16 GEMM operands ----------
__global__ void k_packA(const float* __restrict__ mel, const float* __restrict__ sinb,
                        const float* __restrict__ noise, _Float16* __restrict__ Ah) {
    int i = blockIdx.x * blockDim.x + threadIdx.x;
    if (i >= 2048 * KP_) return;
    int k = i % KP_, row = i / KP_;
    int b = row >> 10, f = row & 1023;
    float v = 0.0f;
    if (k < NMEL_)            v = mel[(size_t)row * NMEL_ + k];
    else if (k < NMEL_ + BLK) v = sinb[(size_t)b * T_ + f * BLK + (k - NMEL_)];
    else if (k < DIN_)        v = noise[(size_t)b * T_ + f * BLK + (k - NMEL_ - BLK)];
    Ah[i] = (_Float16)v;
}

__global__ void k_packW(const float* __restrict__ W, _Float16* __restrict__ Wt) {
    int i = blockIdx.x * blockDim.x + threadIdx.x;
    if (i >= NP_ * KP_) return;
    int k = i % KP_, n = i / KP_;
    float v = (n < DOUT_ && k < DIN_) ? W[(size_t)k * DOUT_ + n] : 0.0f;
    Wt[i] = (_Float16)v;
}

// ---------- 5: control GEMM via v_wmma_f32_16x16x32_f16 ----------
// one wave32 per 16x16 output tile; A (16x32 f16) and B (32x16 f16) fragments
// per ISA layouts: lanes 0-15 hold K 0..7 / 16..23, lanes 16-31 hold K 8..15 / 24..31.
__global__ __launch_bounds__(32) void k_gemm(const _Float16* __restrict__ Ah,
                                             const _Float16* __restrict__ Wt,
                                             const float* __restrict__ bias,
                                             float* __restrict__ ctrl) {
    int lane  = threadIdx.x;
    int tn    = blockIdx.x, tm = blockIdx.y;
    int rlo   = lane & 15, rhalf = lane >> 4;
    const _Float16* arow = Ah + (size_t)(tm * 16 + rlo) * KP_;
    const _Float16* brow = Wt + (size_t)(tn * 16 + rlo) * KP_;
    v8f acc = {};
    for (int k0 = 0; k0 < KP_; k0 += 32) {
        int kb = k0 + rhalf * 8;
        __builtin_prefetch(arow + kb + 64, 0, 0);
        v8h a0 = *(const v8h*)(arow + kb);
        v8h a1 = *(const v8h*)(arow + kb + 16);
        v8h b0 = *(const v8h*)(brow + kb);
        v8h b1 = *(const v8h*)(brow + kb + 16);
        v16h av = __builtin_shufflevector(a0, a1, 0,1,2,3,4,5,6,7,8,9,10,11,12,13,14,15);
        v16h bv = __builtin_shufflevector(b0, b1, 0,1,2,3,4,5,6,7,8,9,10,11,12,13,14,15);
        acc = __builtin_amdgcn_wmma_f32_16x16x32_f16(false, av, false, bv,
                                                     (short)0, acc, false, false);
    }
    int n = tn * 16 + rlo;
    if (n < DOUT_) {
        float bv = bias[n];
        #pragma unroll
        for (int v = 0; v < 8; ++v) {
            int m = tm * 16 + v + rhalf * 8;   // C/D layout: VGPR v -> M=v (+8 upper half)
            ctrl[(size_t)m * DOUT_ + n] = acc[v] + bv;
        }
    }
}

// ---------- 6: harmonic amplitudes ----------
__global__ void k_amps(const float* __restrict__ ctrl, const float* __restrict__ f0,
                       float* __restrict__ amps) {
    int i = blockIdx.x * blockDim.x + threadIdx.x;
    if (i >= 2048 * NH_) return;
    int h = i & 127, row = i >> 7;
    float mask = (f0[row] * (float)(h + 1) < 22050.0f) ? 1.0f : 0.0f;
    mask += 1e-7f;
    amps[i] = expf(ctrl[(size_t)row * DOUT_ + h]) * (1.0f / 128.0f) * mask;
}

// ---------- 7: harmonic bank (writes `sinusoids` output) ----------
__global__ void k_harm(const float* __restrict__ f0, const float* __restrict__ off,
                       const float* __restrict__ amps, float* __restrict__ outsin) {
    int i = blockIdx.x * blockDim.x + threadIdx.x;
    if (i >= BT_) return;
    int b = i >> 19, t = i & (T_ - 1);
    int f = t >> 9, n = t & 511;
    int row = b * NF_ + f;
    float s0 = f0[row] * INV_SR;
    float s1 = (f < NF_ - 1) ? f0[row + 1] * INV_SR : s0;
    float ds = s1 - s0;
    float fn = (float)n;
    float rad = s0 * (fn + 1.0f) + 0.5f * ds * fn * (fn + 1.0f) * (1.0f / 512.0f) + off[row];
    float theta = TWO_PI * rad;
    float w = fn * (1.0f / 512.0f);
    int row2 = b * NF_ + ((f < NF_ - 1) ? f + 1 : f);
    const float* a0 = amps + (size_t)row * NH_;
    const float* a1 = amps + (size_t)row2 * NH_;
    float sum = 0.0f;
    #pragma unroll 8
    for (int h = 0; h < NH_; ++h) {
        float a = a0[h] * (1.0f - w) + a1[h] * w;
        sum += __sinf(theta * (float)(h + 1)) * a;
    }
    outsin[i] = sum;
}

// ---------- 8: STFT + all-pass + inverse FFT + window (per frame) ----------
__global__ __launch_bounds__(256) void k_spec(const float* __restrict__ sinus,
                                              const float* __restrict__ ctrl,
                                              float* __restrict__ ft) {
    int fidx = blockIdx.x;
    int b = fidx / 1025, j = fidx % 1025;
    int tid = threadIdx.x;
    __shared__ float2 buf[WIN_];
    for (int i = tid; i < WIN_; i += 256) {
        int gt = j * BLK + i - 1024;                 // reflect padding
        if (gt < 0) gt = -gt;
        if (gt >= T_) gt = 2 * T_ - 2 - gt;
        float w = hannw(i, 1.0f / 2048.0f);
        buf[__brev((unsigned)i) >> 21] = make_float2(sinus[(size_t)b * T_ + gt] * w, 0.0f);
    }
    __syncthreads();
    fft_lds<WIN_>(buf, tid, 256, -1.0f);
    int jc = (j < NF_) ? j : NF_ - 1;
    const float* hrow = ctrl + (size_t)(b * NF_ + jc) * DOUT_ + NH_;
    float2 vals[5]; int ks[5]; int cnt = 0;
    for (int k = tid; k <= 1024; k += 256) {
        float s, c;
        __sincosf(PI_F * hrow[k], &s, &c);
        vals[cnt] = cmulf(buf[k], make_float2(c, s));
        ks[cnt] = k; ++cnt;
    }
    __syncthreads();
    for (int q = 0; q < cnt; ++q) {
        int k = ks[q];
        buf[k] = vals[q];
        if (k > 0 && k < 1024) buf[WIN_ - k] = make_float2(vals[q].x, -vals[q].y);
    }
    __syncthreads();
    float2 tmp[8];
    for (int i = tid, q = 0; i < WIN_; i += 256, ++q) tmp[q] = buf[__brev((unsigned)i) >> 21];
    __syncthreads();
    for (int i = tid, q = 0; i < WIN_; i += 256, ++q) buf[i] = tmp[q];
    __syncthreads();
    fft_lds<WIN_>(buf, tid, 256, 1.0f);
    for (int i = tid; i < WIN_; i += 256) {
        float w = hannw(i, 1.0f / 2048.0f);
        ft[(size_t)fidx * WIN_ + i] = buf[i].x * (1.0f / 2048.0f) * w;
    }
}

// ---------- 9: ISTFT overlap-add + window-sum normalize ----------
__global__ void k_ola_harm(const float* __restrict__ ft, float* __restrict__ harm) {
    int i = blockIdx.x * blockDim.x + threadIdx.x;
    if (i >= BT_) return;
    int b = i >> 19, t = i & (T_ - 1);
    int tp = t + 1024;
    int lo = tp - 2047;
    int jmin = lo > 0 ? (lo + 511) >> 9 : 0;
    int jmax = tp >> 9; if (jmax > 1024) jmax = 1024;
    float acc = 0.0f, ws = 0.0f;
    for (int j = jmin; j <= jmax; ++j) {
        int o = tp - j * BLK;
        float w = hannw(o, 1.0f / 2048.0f);
        acc += ft[(size_t)(b * 1025 + j) * WIN_ + o];
        ws += w * w;
    }
    harm[i] = acc / (ws > 1e-11f ? ws : 1.0f);
}

// ---------- 10: noise filter: IR build + 1024-pt FFT convolution ----------
__global__ __launch_bounds__(256) void k_noise(const float* __restrict__ noise,
                                               const float* __restrict__ ctrl,
                                               float* __restrict__ yn) {
    int fidx = blockIdx.x;
    int b = fidx >> 10, f = fidx & 1023;
    int tid = threadIdx.x;
    __shared__ float2 buf[1024];
    __shared__ float2 hsp[1024];
    __shared__ float  ir[512];
    const float* mrow = ctrl + (size_t)fidx * DOUT_ + NH_ + BINS_;
    const float* prow = mrow + NMAG_;
    // ir = irfft(exp(m + i*pi*p)/128, n=510) by direct inverse DFT
    for (int s2 = tid; s2 < 510; s2 += 256) {
        float acc = expf(mrow[0]) * __cosf(PI_F * prow[0]);
        for (int k = 1; k < 255; ++k) {
            float m = expf(mrow[k]);
            float sp, cp; __sincosf(PI_F * prow[k], &sp, &cp);
            float tw = TWO_PI * (float)(k * s2) * (1.0f / 510.0f);
            float st, ct; __sincosf(tw, &st, &ct);
            acc += 2.0f * m * (cp * ct - sp * st);
        }
        acc += expf(mrow[255]) * __cosf(PI_F * prow[255]) * ((s2 & 1) ? -1.0f : 1.0f);
        ir[s2] = acc * (1.0f / 510.0f) * (1.0f / 128.0f);
    }
    __syncthreads();
    // roll by 255, Hann(510) window, zero-pad to 1024, FFT
    for (int i = tid; i < 1024; i += 256) {
        float v = 0.0f;
        if (i < 510) {
            int src = i - 255; if (src < 0) src += 510;
            v = ir[src] * hannw(i, 1.0f / 510.0f);
        }
        buf[__brev((unsigned)i) >> 22] = make_float2(v, 0.0f);
    }
    __syncthreads();
    fft_lds<1024>(buf, tid, 256, -1.0f);
    for (int i = tid; i < 1024; i += 256) hsp[i] = buf[i];
    __syncthreads();
    // noise frame, zero-pad to 1024, FFT, multiply, inverse FFT
    for (int i = tid; i < 1024; i += 256) {
        float v = (i < BLK) ? noise[(size_t)b * T_ + f * BLK + i] : 0.0f;
        buf[__brev((unsigned)i) >> 22] = make_float2(v, 0.0f);
    }
    __syncthreads();
    fft_lds<1024>(buf, tid, 256, -1.0f);
    for (int i = tid; i < 1024; i += 256) buf[i] = cmulf(buf[i], hsp[i]);
    __syncthreads();
    float2 tmp[4];
    for (int i = tid, q = 0; i < 1024; i += 256, ++q) tmp[q] = buf[__brev((unsigned)i) >> 22];
    __syncthreads();
    for (int i = tid, q = 0; i < 1024; i += 256, ++q) buf[i] = tmp[q];
    __syncthreads();
    fft_lds<1024>(buf, tid, 256, 1.0f);
    for (int i = tid; i < 1024; i += 256)
        yn[(size_t)fidx * 1024 + i] = buf[i].x * (1.0f / 1024.0f);
}

// ---------- 11: noise overlap-add + final sum ----------
__global__ void k_ola_noise(const float* __restrict__ yn, const float* __restrict__ harm,
                            float* __restrict__ nout, float* __restrict__ sig) {
    int i = blockIdx.x * blockDim.x + threadIdx.x;
    if (i >= BT_) return;
    int b = i >> 19, t = i & (T_ - 1);
    int base = t + 255;
    int lo = base - 1023;
    int jmin = lo > 0 ? (lo + 511) >> 9 : 0;
    int jmax = base >> 9; if (jmax > 1023) jmax = 1023;
    float acc = 0.0f;
    for (int j = jmin; j <= jmax; ++j)
        acc += yn[(size_t)(b * 1024 + j) * 1024 + (base - j * BLK)];
    nout[i] = acc;
    sig[i]  = harm[i] + acc;
}

extern "C" void kernel_launch(void* const* d_in, const int* in_sizes, int n_in,
                              void* d_out, int out_size, void* d_ws, size_t ws_size,
                              hipStream_t stream) {
    const float* mel   = (const float*)d_in[0];
    const float* f0    = (const float*)d_in[1];
    const float* noise = (const float*)d_in[2];
    const float* W     = (const float*)d_in[3];
    const float* bias  = (const float*)d_in[4];

    float* ws = (float*)d_ws;
    float* off  = ws + OFF_OFFS;
    float* sinb = ws + OFF_SINB;
    float* ctrl = ws + OFF_CTRL;
    float* amps = ws + OFF_AMPS;
    float* ft   = ws + OFF_FT;
    float* yn   = ws + OFF_YN;
    _Float16* Ah = (_Float16*)(ws + OFF_AH);
    _Float16* Wt = (_Float16*)(ws + OFF_WH);

    float* o_sig  = (float*)d_out;
    float* o_sin  = o_sig + BT_;
    float* o_harm = o_sin + BT_;
    float* o_nout = o_harm + BT_;

    k_offsets<<<B_, 32, 0, stream>>>(f0, off);
    k_sinbase<<<(BT_ + 255) / 256, 256, 0, stream>>>(f0, off, sinb);
    k_packA<<<(2048 * KP_ + 255) / 256, 256, 0, stream>>>(mel, sinb, noise, Ah);
    k_packW<<<(NP_ * KP_ + 255) / 256, 256, 0, stream>>>(W, Wt);
    k_gemm<<<dim3(NP_ / 16, 2048 / 16), 32, 0, stream>>>(Ah, Wt, bias, ctrl);
    k_amps<<<(2048 * NH_ + 255) / 256, 256, 0, stream>>>(ctrl, f0, amps);
    k_harm<<<(BT_ + 255) / 256, 256, 0, stream>>>(f0, off, amps, o_sin);
    k_spec<<<B_ * 1025, 256, 0, stream>>>(o_sin, ctrl, ft);
    k_ola_harm<<<(BT_ + 255) / 256, 256, 0, stream>>>(ft, o_harm);
    k_noise<<<B_ * 1024, 256, 0, stream>>>(noise, ctrl, yn);
    k_ola_noise<<<(BT_ + 255) / 256, 256, 0, stream>>>(yn, o_harm, o_nout, o_sig);
}
